// AttentionWithDropout_43559558316696
// MI455X (gfx1250) — matrix operand order repaired and checked by reading, not compiled, not run on hardware
//
#include <hip/hip_runtime.h>
#include <hip/hip_bf16.h>
#include <cstdint>

#define DEV __device__ __forceinline__

typedef __attribute__((ext_vector_type(16))) __bf16 v16bf;
typedef __attribute__((ext_vector_type(8))) float v8f;
typedef __attribute__((ext_vector_type(4))) int v4i;

#if defined(__AMDGCN__) && \
    __has_builtin(__builtin_amdgcn_global_load_async_to_lds_b128) && \
    __has_builtin(__builtin_amdgcn_s_wait_asynccnt)
#define HAVE_ASYNC_LDS 1
#endif

namespace {

constexpr int D_MODEL = 1024;
constexpr int N_HEADS = 16;
constexpr int D_K = 64;
constexpr int SEQ = 2048;
constexpr int BATCH = 2;
constexpr int BS = BATCH * SEQ;  // 4096 total rows
constexpr float KEEP = 0.9f;
constexpr float INV_KEEP = 1.0f / KEEP;
constexpr unsigned KEEP_THR24 = (unsigned)(KEEP * 16777216.0f);

// Round-to-nearest-even f32 -> bf16, two at a time packed into a dword.
DEV unsigned pk_bf16x2(float a, float b) {
  unsigned ua = __float_as_uint(a), ub = __float_as_uint(b);
  ua += 0x7FFFu + ((ua >> 16) & 1u);
  ub += 0x7FFFu + ((ub >> 16) & 1u);
  return (ua >> 16) | (ub & 0xFFFF0000u);
}
DEV unsigned short bf16_1(float a) {
  unsigned u = __float_as_uint(a);
  u += 0x7FFFu + ((u >> 16) & 1u);
  return (unsigned short)(u >> 16);
}
// Counter-based hash RNG (deterministic per element, per key).
DEV unsigned hashrng(unsigned idx, unsigned key) {
  unsigned x = idx * 0x9E3779B9u ^ key;
  x ^= x >> 16; x *= 0x7FEB352Du;
  x ^= x >> 15; x *= 0x846CA68Bu;
  x ^= x >> 16;
  return x;
}

// ---- WMMA fragment loaders (ISA 7.12.2 layouts, wave32) ----
// All bf16 fragments are two contiguous 16-byte runs -> 2x b128 loads.
struct Fr { uint4 lo, hi; };
DEV v16bf ld2(const unsigned short* p0, const unsigned short* p1) {
  Fr f;
  f.lo = *reinterpret_cast<const uint4*>(p0);
  f.hi = *reinterpret_cast<const uint4*>(p1);
  return __builtin_bit_cast(v16bf, f);
}
// A fragment 16x32 (MxK): lane M=lane&15; halfs [k0+hi*8 .. +7] and [+16 .. +23]
DEV v16bf load_a_bf16(const unsigned short* base, int row, int ld, int k0, int hi) {
  const unsigned short* p = base + (size_t)row * ld + k0 + (hi << 3);
  return ld2(p, p + 16);
}
// B fragment 32x16 (KxN) from B^T rows (row=N, contiguous K):
// lane N=lane&15; halfs [k0+hi*8 .. +7] and [+16 .. +23]
DEV v16bf load_bt_bf16(const unsigned short* base, int row, int ld, int k0, int hi) {
  const unsigned short* p = base + (size_t)row * ld + k0 + (hi << 3);
  return ld2(p, p + 16);
}

DEV v8f wmma_bf16(v16bf a, v16bf b, v8f c) {
  return __builtin_amdgcn_wmma_f32_16x16x32_bf16(
      /*neg_a=*/false, a, /*neg_b=*/false, b,
      /*c_mod=*/(short)0, c, /*reuse_a=*/false, /*reuse_b=*/false);
}

#if defined(HAVE_ASYNC_LDS)
typedef __attribute__((address_space(1))) v4i as1_v4i;
typedef __attribute__((address_space(3))) v4i as3_v4i;
// One 16-byte ASYNCcnt-tracked DMA: global -> LDS (no VGPR data path).
DEV void async_cp16(const unsigned short* gsrc, unsigned short* ldst) {
  __builtin_amdgcn_global_load_async_to_lds_b128(
      (as1_v4i*)(void*)const_cast<unsigned short*>(gsrc),
      (as3_v4i*)(void*)ldst, 0, 0);
}
#endif

}  // namespace

// ---------------------------------------------------------------------------
// Kernel 0: elementwise f32 -> bf16 conversion (memory bound).
// ---------------------------------------------------------------------------
__global__ __launch_bounds__(256) void cvt_bf16_kernel(
    const float* __restrict__ src, unsigned* __restrict__ dst, int n2) {
  int i = blockIdx.x * 256 + threadIdx.x;
  if (i < n2) {
    float2 t = reinterpret_cast<const float2*>(src)[i];
    dst[i] = pk_bf16x2(t.x, t.y);
  }
}

// ---------------------------------------------------------------------------
// Kernel 1: Q/K/V projections. y = x @ W.T + b  (all-bf16 operands).
// Each wave computes a 32x32 C block: 4 WMMAs per 8 b128 loads.
// Q,K -> (B,H,S,Dk); V -> transposed (B,H,Dk,S).
// grid = (BS/32, D_MODEL/128, 3), block = 128 (4 waves)
// ---------------------------------------------------------------------------
__global__ __launch_bounds__(128) void qkv_proj_kernel(
    const unsigned short* __restrict__ xb,
    const unsigned short* __restrict__ Wqb, const float* __restrict__ bq,
    const unsigned short* __restrict__ Wkb, const float* __restrict__ bk,
    const unsigned short* __restrict__ Wvb, const float* __restrict__ bv,
    unsigned short* __restrict__ Qb, unsigned short* __restrict__ Kb,
    unsigned short* __restrict__ Vtb) {
  const int lane = threadIdx.x & 31, lh = lane & 15, hi = lane >> 4;
  const int w = threadIdx.x >> 5;
  const int mbase = blockIdx.x * 32;
  const int nbase = blockIdx.y * 128 + w * 32;
  const int which = blockIdx.z;
  const unsigned short* W = (which == 0) ? Wqb : (which == 1) ? Wkb : Wvb;
  const float* bias = (which == 0) ? bq : (which == 1) ? bk : bv;

  v8f acc[2][2];
  const v8f vzero = {};
  acc[0][0] = vzero; acc[0][1] = vzero; acc[1][0] = vzero; acc[1][1] = vzero;
  const int ar0 = mbase + lh, ar1 = ar0 + 16;
  const int nr0 = nbase + lh, nr1 = nr0 + 16;

#pragma unroll 4
  for (int k0 = 0; k0 < D_MODEL; k0 += 32) {
    // near-cache (WGP-scope) prefetch two k-slabs ahead
    __builtin_prefetch(xb + (size_t)ar0 * D_MODEL + k0 + 64, 0, 3);
    __builtin_prefetch(W + (size_t)nr0 * D_MODEL + k0 + 64, 0, 3);
    v16bf a0 = load_a_bf16(xb, ar0, D_MODEL, k0, hi);
    v16bf a1 = load_a_bf16(xb, ar1, D_MODEL, k0, hi);
    v16bf b0 = load_bt_bf16(W, nr0, D_MODEL, k0, hi);
    v16bf b1 = load_bt_bf16(W, nr1, D_MODEL, k0, hi);
    acc[0][0] = wmma_bf16(a0, b0, acc[0][0]);
    acc[0][1] = wmma_bf16(a0, b1, acc[0][1]);
    acc[1][0] = wmma_bf16(a1, b0, acc[1][0]);
    acc[1][1] = wmma_bf16(a1, b1, acc[1][1]);
  }
#pragma unroll
  for (int ni = 0; ni < 2; ++ni) {
    const int n = nbase + ni * 16 + lh;
    const float bval = bias[n];
    const int h = n >> 6, dk = n & (D_K - 1);
#pragma unroll
    for (int mi = 0; mi < 2; ++mi) {
#pragma unroll
      for (int r = 0; r < 8; ++r) {
        int m = mbase + mi * 16 + r + hi * 8;   // global row in (B*S)
        int b_ = m >> 11, s = m & (SEQ - 1);
        unsigned short o = bf16_1(acc[mi][ni][r] + bval);
        if (which == 2) {
          Vtb[(((size_t)(b_ * N_HEADS + h)) * D_K + dk) * SEQ + s] = o;
        } else {
          size_t addr = (((size_t)(b_ * N_HEADS + h)) * SEQ + s) * D_K + dk;
          if (which == 0) Qb[addr] = o; else Kb[addr] = o;
        }
      }
    }
  }
}

// ---------------------------------------------------------------------------
// Kernel 2: flash attention with online softmax + dropout on P.
// Block = 128 threads (4 waves x 16 query rows, same (b,h)). K/V tiles for
// the 32-key slab are staged once per block into double-buffered LDS via
// GLOBAL_LOAD_ASYNC_TO_LDS_B128 (ASYNCcnt), overlapping DMA of tile j+1 with
// WMMA+softmax of tile j. Falls back to direct b128 loads without the builtin.
// grid = (SEQ/64, B*H)
// ---------------------------------------------------------------------------
__global__ __launch_bounds__(128) void flash_attn_kernel(
    const unsigned short* __restrict__ Qb, const unsigned short* __restrict__ Kb,
    const unsigned short* __restrict__ Vtb, unsigned short* __restrict__ Ob) {
  __shared__ alignas(16) unsigned short plds[4][16][32];  // per-wave P tile
#if defined(HAVE_ASYNC_LDS)
  __shared__ alignas(16) unsigned short kbuf[2][32][D_K];   // 2 x 4 KiB
  __shared__ alignas(16) unsigned short vbuf[2][D_K][32];   // 2 x 4 KiB
#endif
  const int t = threadIdx.x;
  const int lane = t & 31, lh = lane & 15, hi = lane >> 4;
  const int w = t >> 5;
  const int bh = blockIdx.y;                  // b*H + h
  const int m0 = blockIdx.x * 64 + w * 16;    // query tile base

  const unsigned short* q  = Qb + (size_t)bh * SEQ * D_K;
  const unsigned short* kp = Kb + (size_t)bh * SEQ * D_K;
  const unsigned short* vt = Vtb + (size_t)bh * D_K * SEQ;

  const v16bf qa0 = load_a_bf16(q, m0 + lh, D_K, 0, hi);
  const v16bf qa1 = load_a_bf16(q, m0 + lh, D_K, 32, hi);

  float mrow[8], lrow[8];
  v8f oacc[4];
  const v8f vzero = {};
#pragma unroll
  for (int r = 0; r < 8; ++r) { mrow[r] = -3.0e38f; lrow[r] = 0.f; }
#pragma unroll
  for (int dt = 0; dt < 4; ++dt) oacc[dt] = vzero;

  const unsigned rowkey = (unsigned)bh * (unsigned)SEQ;

#if defined(HAVE_ASYNC_LDS)
  // Stage K (32 keys x 64 dk, contiguous 4 KiB) and V^T slab (64 d x 32 keys):
  // 4 async b128s per thread per tile.
  auto stage = [&](int buf, int j) {
    const unsigned short* ks = kp + (size_t)j * D_K;
    unsigned short* kd = &kbuf[buf][0][0];
    async_cp16(ks + t * 8, kd + t * 8);
    async_cp16(ks + 1024 + t * 8, kd + 1024 + t * 8);
    const int r0 = t >> 2, o0 = (t & 3) * 8;
    const int c1 = t + 128, r1 = c1 >> 2, o1 = (c1 & 3) * 8;
    async_cp16(vt + (size_t)r0 * SEQ + j + o0, &vbuf[buf][r0][o0]);
    async_cp16(vt + (size_t)r1 * SEQ + j + o1, &vbuf[buf][r1][o1]);
  };
  stage(0, 0);
#endif

  for (int jp = 0; jp < SEQ / 32; ++jp) {
    const int j0 = jp * 32;
#if defined(HAVE_ASYNC_LDS)
    const int cur = jp & 1;
    if (jp + 1 < SEQ / 32) {
      stage(cur ^ 1, j0 + 32);                 // DMA next tile
      __builtin_amdgcn_s_wait_asynccnt(4);     // current tile's 4 DMAs done
    } else {
      __builtin_amdgcn_s_wait_asynccnt(0);
    }
    __syncthreads();                           // tile visible block-wide
    const unsigned short* kt = &kbuf[cur][0][0];  // local keys, ld = 64
    const unsigned short* vtl = &vbuf[cur][0][0]; // local V^T,  ld = 32
    const int jl = 0;                          // key base within LDS tile
    const int vld = 32;
#else
    const unsigned short* kt = kp;             // global, ld = 64
    const unsigned short* vtl = vt;            // global, ld = SEQ
    const int jl = j0;
    const int vld = SEQ;
#endif
    // ---- scores: Q(16x64) @ K^T(64x32), two 16x16 tiles ----
    v8f s0 = vzero, s1 = vzero;
    {
      v16bf kb0 = load_bt_bf16(kt, jl + lh, D_K, 0, hi);
      v16bf kb1 = load_bt_bf16(kt, jl + lh, D_K, 32, hi);
      s0 = wmma_bf16(qa0, kb0, s0);
      s0 = wmma_bf16(qa1, kb1, s0);
    }
    {
      v16bf kb0 = load_bt_bf16(kt, jl + 16 + lh, D_K, 0, hi);
      v16bf kb1 = load_bt_bf16(kt, jl + 16 + lh, D_K, 32, hi);
      s1 = wmma_bf16(qa0, kb0, s1);
      s1 = wmma_bf16(qa1, kb1, s1);
    }
    // ---- online softmax over the 32 new columns (row = VGPR idx) ----
#pragma unroll
    for (int r = 0; r < 8; ++r) {
      float a = s0[r] * 0.125f;          // 1/sqrt(64)
      float b = s1[r] * 0.125f;
      float mt_ = fmaxf(a, b);
#pragma unroll
      for (int mm = 8; mm >= 1; mm >>= 1) mt_ = fmaxf(mt_, __shfl_xor(mt_, mm, 16));
      float mnew = fmaxf(mrow[r], mt_);
      float alpha = __expf(mrow[r] - mnew);
      mrow[r] = mnew;
      float p0 = __expf(a - mnew), p1 = __expf(b - mnew);
      float ps = p0 + p1;
#pragma unroll
      for (int mm = 8; mm >= 1; mm >>= 1) ps += __shfl_xor(ps, mm, 16);
      lrow[r] = lrow[r] * alpha + ps;
#pragma unroll
      for (int dt = 0; dt < 4; ++dt) oacc[dt][r] *= alpha;
      // dropout on P (mask indexed like attn[B,H,S,S]), write bf16 to LDS
      const int row = r + hi * 8;
      const unsigned base = (rowkey + (unsigned)(m0 + row)) * (unsigned)SEQ;
      unsigned r0 = hashrng(base + (unsigned)(j0 + lh), 0x243F6A88u);
      unsigned r1 = hashrng(base + (unsigned)(j0 + 16 + lh), 0x243F6A88u);
      float q0 = ((r0 & 0xFFFFFFu) < KEEP_THR24) ? p0 * INV_KEEP : 0.f;
      float q1 = ((r1 & 0xFFFFFFu) < KEEP_THR24) ? p1 * INV_KEEP : 0.f;
      plds[w][row][lh] = bf16_1(q0);
      plds[w][row][16 + lh] = bf16_1(q1);
    }
    // ---- reshape P (C-layout) -> A-fragment via LDS (2x ds_load_b128) ----
    const unsigned short* pp = &plds[w][lh][hi << 3];
    v16bf pa = ld2(pp, pp + 16);
    // ---- O += P(16x32) @ V(32x64); V^T rows contiguous in keys ----
#pragma unroll
    for (int dt = 0; dt < 4; ++dt) {
      const unsigned short* vrow = vtl + (size_t)(dt * 16 + lh) * vld + jl + (hi << 3);
      oacc[dt] = wmma_bf16(pa, ld2(vrow, vrow + 16), oacc[dt]);
    }
#if defined(HAVE_ASYNC_LDS)
    __syncthreads();   // everyone done reading buf `cur` before it is re-DMA'd
#endif
  }
  // ---- epilogue: divide by l, store to (B,S,D_MODEL) bf16 ----
  const int h = bh & (N_HEADS - 1), b_ = bh >> 4;
#pragma unroll
  for (int dt = 0; dt < 4; ++dt) {
#pragma unroll
    for (int r = 0; r < 8; ++r) {
      int s = m0 + r + hi * 8;
      int col = h * D_K + dt * 16 + lh;
      float val = oacc[dt][r] / lrow[r];
      Ob[((size_t)(b_ * SEQ + s)) * D_MODEL + col] = bf16_1(val);
    }
  }
}

// ---------------------------------------------------------------------------
// Kernel 3: out = dropout(O @ Wo.T + bo), f32 output. 32x32 block per wave.
// grid = (BS/32, D_MODEL/128), block = 128
// ---------------------------------------------------------------------------
__global__ __launch_bounds__(128) void out_proj_kernel(
    const unsigned short* __restrict__ Ob, const unsigned short* __restrict__ Wob,
    const float* __restrict__ bo, float* __restrict__ out) {
  const int lane = threadIdx.x & 31, lh = lane & 15, hi = lane >> 4;
  const int w = threadIdx.x >> 5;
  const int mbase = blockIdx.x * 32;
  const int nbase = blockIdx.y * 128 + w * 32;

  v8f acc[2][2];
  const v8f vzero = {};
  acc[0][0] = vzero; acc[0][1] = vzero; acc[1][0] = vzero; acc[1][1] = vzero;
  const int ar0 = mbase + lh, ar1 = ar0 + 16;
  const int nr0 = nbase + lh, nr1 = nr0 + 16;

#pragma unroll 4
  for (int k0 = 0; k0 < D_MODEL; k0 += 32) {
    __builtin_prefetch(Ob + (size_t)ar0 * D_MODEL + k0 + 64, 0, 3);
    __builtin_prefetch(Wob + (size_t)nr0 * D_MODEL + k0 + 64, 0, 3);
    v16bf a0 = load_a_bf16(Ob, ar0, D_MODEL, k0, hi);
    v16bf a1 = load_a_bf16(Ob, ar1, D_MODEL, k0, hi);
    v16bf b0 = load_bt_bf16(Wob, nr0, D_MODEL, k0, hi);
    v16bf b1 = load_bt_bf16(Wob, nr1, D_MODEL, k0, hi);
    acc[0][0] = wmma_bf16(a0, b0, acc[0][0]);
    acc[0][1] = wmma_bf16(a0, b1, acc[0][1]);
    acc[1][0] = wmma_bf16(a1, b0, acc[1][0]);
    acc[1][1] = wmma_bf16(a1, b1, acc[1][1]);
  }
#pragma unroll
  for (int ni = 0; ni < 2; ++ni) {
    const int n = nbase + ni * 16 + lh;
    const float bval = bo[n];
#pragma unroll
    for (int mi = 0; mi < 2; ++mi) {
#pragma unroll
      for (int r = 0; r < 8; ++r) {
        int m = mbase + mi * 16 + r + hi * 8;
        float val = acc[mi][ni][r] + bval;
        unsigned idx = (unsigned)m * (unsigned)D_MODEL + (unsigned)n;
        unsigned rr = hashrng(idx, 0x13198A2Eu);
        val = ((rr & 0xFFFFFFu) < KEEP_THR24) ? val * INV_KEEP : 0.f;
        out[(size_t)m * D_MODEL + n] = val;
      }
    }
  }
}

// ---------------------------------------------------------------------------
extern "C" void kernel_launch(void* const* d_in, const int* in_sizes, int n_in,
                              void* d_out, int out_size, void* d_ws, size_t ws_size,
                              hipStream_t stream) {
  (void)in_sizes; (void)n_in; (void)out_size; (void)ws_size;
  const float* x  = (const float*)d_in[0];
  const float* Wq = (const float*)d_in[1];
  const float* bq = (const float*)d_in[2];
  const float* Wk = (const float*)d_in[3];
  const float* bk = (const float*)d_in[4];
  const float* Wv = (const float*)d_in[5];
  const float* bv = (const float*)d_in[6];
  const float* Wo = (const float*)d_in[7];
  const float* bo = (const float*)d_in[8];
  float* out = (float*)d_out;

  char* ws = (char*)d_ws;
  const size_t MB = 1024 * 1024;
  unsigned short* xb  = (unsigned short*)(ws);            //  8 MiB (4096x1024)
  unsigned short* Wqb = (unsigned short*)(ws +  8 * MB);  //  2 MiB each
  unsigned short* Wkb = (unsigned short*)(ws + 10 * MB);
  unsigned short* Wvb = (unsigned short*)(ws + 12 * MB);
  unsigned short* Wob = (unsigned short*)(ws + 14 * MB);
  unsigned short* Qb  = (unsigned short*)(ws + 16 * MB);  //  8 MiB each
  unsigned short* Kb  = (unsigned short*)(ws + 24 * MB);
  unsigned short* Vtb = (unsigned short*)(ws + 32 * MB);
  unsigned short* Ob  = (unsigned short*)(ws + 40 * MB);  // total 48 MiB

  const int nx2 = BS * D_MODEL / 2;       // 2M float2 pairs
  const int nw2 = D_MODEL * D_MODEL / 2;  // 512K
  cvt_bf16_kernel<<<dim3((nx2 + 255) / 256), dim3(256), 0, stream>>>(x, (unsigned*)xb, nx2);
  cvt_bf16_kernel<<<dim3((nw2 + 255) / 256), dim3(256), 0, stream>>>(Wq, (unsigned*)Wqb, nw2);
  cvt_bf16_kernel<<<dim3((nw2 + 255) / 256), dim3(256), 0, stream>>>(Wk, (unsigned*)Wkb, nw2);
  cvt_bf16_kernel<<<dim3((nw2 + 255) / 256), dim3(256), 0, stream>>>(Wv, (unsigned*)Wvb, nw2);
  cvt_bf16_kernel<<<dim3((nw2 + 255) / 256), dim3(256), 0, stream>>>(Wo, (unsigned*)Wob, nw2);

  dim3 b128(128, 1, 1);
  qkv_proj_kernel<<<dim3(BS / 32, D_MODEL / 128, 3), b128, 0, stream>>>(
      xb, Wqb, bq, Wkb, bk, Wvb, bv, Qb, Kb, Vtb);
  flash_attn_kernel<<<dim3(SEQ / 64, BATCH * N_HEADS, 1), b128, 0, stream>>>(
      Qb, Kb, Vtb, Ob);
  out_proj_kernel<<<dim3(BS / 32, D_MODEL / 128, 1), b128, 0, stream>>>(
      Ob, Wob, bo, out);
}
